// DecoderRNN_52956946760182
// MI455X (gfx1250) — matrix operand order repaired and checked by reading, not compiled, not run on hardware
//
#include <hip/hip_runtime.h>

// ---------------------------------------------------------------------------
// DecoderRNN on MI455X (gfx1250): bf16 WMMA GEMMs, fused GRU epilogue,
// async global->LDS DMA (ASYNCcnt) with double-buffered software pipeline.
// Weights (28MB bf16) stay resident in the 192MB L2 across all 255 steps.
// ---------------------------------------------------------------------------

typedef __attribute__((ext_vector_type(16))) __bf16 v16bf;
typedef __attribute__((ext_vector_type(8)))  __bf16 v8bf;
typedef __attribute__((ext_vector_type(8)))  float  v8f;

struct alignas(16) U16x8 { unsigned short u[8]; };

constexpr int B  = 512;
constexpr int S  = 256;
constexpr int E  = 512;
constexpr int Z  = 512;
constexpr int H  = 1024;
constexpr int V  = 512;
constexpr int K0 = E + Z;   // 1024
constexpr int H3 = 3 * H;   // 3072

__device__ __forceinline__ float bf2f(unsigned short u) {
    return __uint_as_float(((unsigned)u) << 16);
}
__device__ __forceinline__ unsigned short f2bf(float f) {
    unsigned u = __float_as_uint(f);
    return (unsigned short)((u + 0x7FFFu + ((u >> 16) & 1u)) >> 16);
}
__device__ __forceinline__ float sigmoidf_(float x) {
    return 1.0f / (1.0f + __expf(-x));
}
__device__ __forceinline__ v16bf join16(v8bf lo, v8bf hi) {
    return __builtin_shufflevector(lo, hi, 0,1,2,3,4,5,6,7,8,9,10,11,12,13,14,15);
}

// Async DMA: global -> LDS, 16B per lane, tracked by ASYNCcnt (gfx1250).
// LDS dest address comes from VGPR (low 32 bits of the generic shared ptr =
// LDS offset per the flat->LDS aperture mapping).
__device__ __forceinline__ void async_ld_b128(const unsigned short* g,
                                              unsigned short* l) {
    unsigned loff = (unsigned)(unsigned long long)(size_t)l;
    unsigned long long ga = (unsigned long long)(size_t)g;
    asm volatile("global_load_async_to_lds_b128 %0, %1, off"
                 :: "v"(loff), "v"(ga) : "memory");
}
#define WAIT_ASYNC_LE_4() asm volatile("s_wait_asynccnt 4" ::: "memory")
#define WAIT_ASYNC_LE_2() asm volatile("s_wait_asynccnt 2" ::: "memory")
#define WAIT_ASYNC_0()    asm volatile("s_wait_asynccnt 0" ::: "memory")

// ---------------------------------------------------------------------------
// Fused GRU layer: h_new = GRU(inp, h_prev). 64x64 [batch x hidden] tile per
// workgroup, 8 wave32s (4M x 2N), 16x32 per wave. r,z gates merge input+hidden
// GEMMs into one accumulator; n keeps them split (r gates only hidden part).
// 64-tile pipeline: phase 0 = input path (Wih), phase 1 = hidden path (Whh).
// ---------------------------------------------------------------------------
__global__ __launch_bounds__(256)
void gru_layer_kernel(const unsigned short* __restrict__ inp,    // [B,1024] bf16
                      const unsigned short* __restrict__ hprev,  // [B,H]    bf16
                      const unsigned short* __restrict__ Wih,    // [3H,1024] bf16
                      const unsigned short* __restrict__ Whh,    // [3H,H]    bf16
                      const float* __restrict__ bih,             // [3H]
                      const float* __restrict__ bhh,             // [3H]
                      unsigned short* __restrict__ hnew)         // [B,H] bf16
{
    constexpr int KD = 1024;
    constexpr int NT = 64;                        // 2 phases x 32 k-tiles
    __shared__ unsigned short sA[2][64][32];      //  8 KB double-buffered act
    __shared__ unsigned short sW[2][3][64][32];   // 24 KB double-buffered wgt

    const int tid   = threadIdx.x;
    const int lane  = tid & 31;
    const int wid   = tid >> 5;
    const int lm    = lane & 15;
    const int lk    = lane >> 4;
    const int mOff  = (wid >> 1) * 16;
    const int nOff  = (wid & 1) * 32;
    const int rowBase = blockIdx.y * 64;          // batch tile
    const int nBase   = blockIdx.x * 64;          // hidden tile

    const int sr = tid >> 2;                      // staging row (0..63)
    const int sc = (tid & 3) * 8;                 // staging col chunk (16B)

    const v8f vzero = {0.f,0.f,0.f,0.f,0.f,0.f,0.f,0.f};
    v8f accR[2]  = {vzero, vzero};
    v8f accZg[2] = {vzero, vzero};
    v8f accNI[2] = {vzero, vzero};
    v8f accNH[2] = {vzero, vzero};

    // issue the 4 async DMAs (1 act + 3 gate-weight 16B chunks) for tile t
    auto issue = [&](int t) {
        const int p  = t >> 5;
        const int k0 = (t & 31) * 32;
        const int bi = t & 1;
        const unsigned short* Ap = p ? hprev : inp;
        const unsigned short* Wp = p ? Whh   : Wih;
        async_ld_b128(&Ap[(size_t)(rowBase + sr) * KD + k0 + sc], &sA[bi][sr][sc]);
        #pragma unroll
        for (int g = 0; g < 3; ++g)
            async_ld_b128(&Wp[(size_t)(g * H + nBase + sr) * KD + k0 + sc],
                          &sW[bi][g][sr][sc]);
    };

    // one tile of WMMAs: 2 N-subtiles x {r, z, n} (n accum chosen per phase)
#define GRU_COMPUTE_TILE(BI, ACCN)                                             \
    do {                                                                       \
        v16bf afrag = join16(*(const v8bf*)&sA[BI][mOff + lm][lk * 16],        \
                             *(const v8bf*)&sA[BI][mOff + lm][lk * 16 + 8]);   \
        _Pragma("unroll")                                                      \
        for (int s = 0; s < 2; ++s) {                                          \
            const unsigned short* bp0 = &sW[BI][0][nOff + s * 16 + lm][lk*16]; \
            const unsigned short* bp1 = &sW[BI][1][nOff + s * 16 + lm][lk*16]; \
            const unsigned short* bp2 = &sW[BI][2][nOff + s * 16 + lm][lk*16]; \
            v16bf bf0 = join16(*(const v8bf*)bp0, *(const v8bf*)(bp0 + 8));    \
            v16bf bf1 = join16(*(const v8bf*)bp1, *(const v8bf*)(bp1 + 8));    \
            v16bf bf2 = join16(*(const v8bf*)bp2, *(const v8bf*)(bp2 + 8));    \
            accR[s]  = __builtin_amdgcn_wmma_f32_16x16x32_bf16(                \
                           false, afrag, false, bf0, (short)0, accR[s],        \
                           false, false);                                      \
            accZg[s] = __builtin_amdgcn_wmma_f32_16x16x32_bf16(                \
                           false, afrag, false, bf1, (short)0, accZg[s],       \
                           false, false);                                      \
            ACCN[s]  = __builtin_amdgcn_wmma_f32_16x16x32_bf16(                \
                           false, afrag, false, bf2, (short)0, ACCN[s],        \
                           false, false);                                      \
        }                                                                      \
    } while (0)

    issue(0);
    // phase 0: tiles 0..31 (input path; n-gate -> accNI)
    for (int t = 0; t < 32; ++t) {
        const int bi = t & 1;
        issue(t + 1);                 // t+1 <= 32 < NT always in this loop
        WAIT_ASYNC_LE_4();            // current tile's 4 DMAs complete (in-order)
        __syncthreads();
        GRU_COMPUTE_TILE(bi, accNI);
        __syncthreads();              // safe to re-DMA this buffer next+1 iter
    }
    // phase 1: tiles 32..63 (hidden path; n-gate -> accNH)
    for (int t = 32; t < NT; ++t) {
        const int bi = t & 1;
        if (t + 1 < NT) { issue(t + 1); WAIT_ASYNC_LE_4(); }
        else            { WAIT_ASYNC_0(); }
        __syncthreads();
        GRU_COMPUTE_TILE(bi, accNH);
        __syncthreads();
    }
#undef GRU_COMPUTE_TILE

    // ---- fused GRU epilogue (C layout: VGPR r -> M = r + 8*lk, N = lm) ----
    #pragma unroll
    for (int s = 0; s < 2; ++s) {
        const int ng  = nBase + nOff + s * 16 + lm;
        const float br = bih[ng]         + bhh[ng];
        const float bz = bih[H + ng]     + bhh[H + ng];
        const float bi = bih[2 * H + ng];
        const float bh = bhh[2 * H + ng];
        #pragma unroll
        for (int r = 0; r < 8; ++r) {
            const int mg = rowBase + mOff + lk * 8 + r;
            const float rg = sigmoidf_(accR[s][r]  + br);
            const float zg = sigmoidf_(accZg[s][r] + bz);
            const float nn = tanhf(accNI[s][r] + bi + rg * (accNH[s][r] + bh));
            const float hp = bf2f(hprev[(size_t)mg * H + ng]);
            hnew[(size_t)mg * H + ng] = f2bf((1.0f - zg) * nn + zg * hp);
        }
    }
}

// ---------------------------------------------------------------------------
// Generic WMMA GEMM: C[M,N] = A[M,K] * W[N,K]^T + bias, async double-buffered.
// OUT_BF16: bf16 (hidden init) vs f32 (logits straight into d_out).
// ---------------------------------------------------------------------------
template <bool OUT_BF16>
__global__ __launch_bounds__(256)
void wmma_gemm_kernel(const unsigned short* __restrict__ A, int lda,
                      const unsigned short* __restrict__ W, int ldw,
                      const float* __restrict__ bias,
                      void* __restrict__ outv, int ldo, int K)
{
    __shared__ unsigned short sA[2][64][32];
    __shared__ unsigned short sW[2][64][32];

    const int tid  = threadIdx.x;
    const int lane = tid & 31;
    const int wid  = tid >> 5;
    const int lm   = lane & 15;
    const int lk   = lane >> 4;
    const int mOff = (wid >> 1) * 16;
    const int nOff = (wid & 1) * 32;
    const int rowBase = blockIdx.y * 64;
    const int nBase   = blockIdx.x * 64;
    const int sr = tid >> 2;
    const int sc = (tid & 3) * 8;

    const v8f vzero = {0.f,0.f,0.f,0.f,0.f,0.f,0.f,0.f};
    v8f acc[2] = {vzero, vzero};

    const int NT = K >> 5;
    auto issue = [&](int t) {
        const int k0 = t * 32;
        const int bi = t & 1;
        async_ld_b128(&A[(size_t)(rowBase + sr) * lda + k0 + sc], &sA[bi][sr][sc]);
        async_ld_b128(&W[(size_t)(nBase + sr) * ldw + k0 + sc],   &sW[bi][sr][sc]);
    };

    issue(0);
    for (int t = 0; t < NT; ++t) {
        const int bi = t & 1;
        if (t + 1 < NT) { issue(t + 1); WAIT_ASYNC_LE_2(); }
        else            { WAIT_ASYNC_0(); }
        __syncthreads();

        v16bf afrag = join16(*(const v8bf*)&sA[bi][mOff + lm][lk * 16],
                             *(const v8bf*)&sA[bi][mOff + lm][lk * 16 + 8]);
        #pragma unroll
        for (int s = 0; s < 2; ++s) {
            const unsigned short* bp = &sW[bi][nOff + s * 16 + lm][lk * 16];
            v16bf bfrag = join16(*(const v8bf*)bp, *(const v8bf*)(bp + 8));
            acc[s] = __builtin_amdgcn_wmma_f32_16x16x32_bf16(
                         false, afrag, false, bfrag, (short)0, acc[s], false, false);
        }
        __syncthreads();
    }

    #pragma unroll
    for (int s = 0; s < 2; ++s) {
        const int ng = nBase + nOff + s * 16 + lm;
        const float bv = bias[ng];
        #pragma unroll
        for (int r = 0; r < 8; ++r) {
            const int mg = rowBase + mOff + lk * 8 + r;
            const float v = acc[s][r] + bv;
            if (OUT_BF16)
                ((unsigned short*)outv)[(size_t)mg * ldo + ng] = f2bf(v);
            else
                ((float*)outv)[(size_t)mg * ldo + ng] = v;
        }
    }
}

// ---------------------------------------------------------------------------
// Small helper kernels
// ---------------------------------------------------------------------------
__global__ void f32_to_bf16_kernel(const float* __restrict__ in,
                                   unsigned short* __restrict__ out, int n) {
    for (int i = blockIdx.x * blockDim.x + threadIdx.x; i < n;
         i += gridDim.x * blockDim.x)
        out[i] = f2bf(in[i]);
}

__global__ void zero_f32_kernel(float* __restrict__ p, int n) {
    for (int i = blockIdx.x * blockDim.x + threadIdx.x; i < n;
         i += gridDim.x * blockDim.x)
        p[i] = 0.0f;
}

__global__ void tok_init_kernel(const int* __restrict__ x, int* __restrict__ tok) {
    int b = blockIdx.x * blockDim.x + threadIdx.x;
    if (b < B) tok[b] = x[b];   // x[0, b]: first token, teacher-forced
}

__global__ void concat_z_kernel(const unsigned short* __restrict__ zbf,
                                unsigned short* __restrict__ inp0) {
    int b = blockIdx.x;
    const U16x8* src = (const U16x8*)(zbf + (size_t)b * Z);
    U16x8* dst = (U16x8*)(inp0 + (size_t)b * K0 + E);
    for (int i = threadIdx.x; i < Z / 8; i += blockDim.x) dst[i] = src[i];
}

__global__ void embed_kernel(const int* __restrict__ tok,
                             const unsigned short* __restrict__ embb,
                             unsigned short* __restrict__ inp0) {
    int b = blockIdx.x;
    int t = tok[b];
    const U16x8* src = (const U16x8*)(embb + (size_t)t * E);
    U16x8* dst = (U16x8*)(inp0 + (size_t)b * K0);
    for (int i = threadIdx.x; i < E / 8; i += blockDim.x) dst[i] = src[i];
}

__global__ __launch_bounds__(256)
void argmax_kernel(const float* __restrict__ logits, int* __restrict__ tok) {
    __shared__ float sv[256];
    __shared__ int   si[256];
    const int b = blockIdx.x, t = threadIdx.x;
    float best = -3.4e38f; int bi = 0;
    for (int v = t; v < V; v += 256) {
        float x = logits[(size_t)b * V + v];
        if (x > best) { best = x; bi = v; }
    }
    sv[t] = best; si[t] = bi;
    __syncthreads();
    for (int off = 128; off > 0; off >>= 1) {
        if (t < off) {
            float xo = sv[t + off]; int io = si[t + off];
            if (xo > sv[t] || (xo == sv[t] && io < si[t])) { sv[t] = xo; si[t] = io; }
        }
        __syncthreads();
    }
    if (t == 0) tok[b] = si[0];
}

// ---------------------------------------------------------------------------
// Host launcher
// ---------------------------------------------------------------------------
extern "C" void kernel_launch(void* const* d_in, const int* in_sizes, int n_in,
                              void* d_out, int out_size, void* d_ws, size_t ws_size,
                              hipStream_t stream) {
    const float* z     = (const float*)d_in[0];
    const int*   x     = (const int*)d_in[1];
    const float* emb   = (const float*)d_in[2];
    const float* W_hid = (const float*)d_in[3];
    const float* b_hid = (const float*)d_in[4];
    const float* W_ih0 = (const float*)d_in[5];
    const float* W_hh0 = (const float*)d_in[6];
    const float* b_ih0 = (const float*)d_in[7];
    const float* b_hh0 = (const float*)d_in[8];
    const float* W_ih1 = (const float*)d_in[9];
    const float* W_hh1 = (const float*)d_in[10];
    const float* b_ih1 = (const float*)d_in[11];
    const float* b_hh1 = (const float*)d_in[12];
    const float* W_out = (const float*)d_in[13];
    const float* b_out = (const float*)d_in[14];
    float* out = (float*)d_out;

    // workspace carve-out (256B aligned)
    char* ws = (char*)d_ws;
    size_t off = 0;
    auto alloc = [&](size_t bytes) -> void* {
        void* p = ws + off;
        off = (off + bytes + 255) & ~(size_t)255;
        return p;
    };
    unsigned short* wih0b = (unsigned short*)alloc((size_t)H3 * K0 * 2);
    unsigned short* whh0b = (unsigned short*)alloc((size_t)H3 * H  * 2);
    unsigned short* wih1b = (unsigned short*)alloc((size_t)H3 * H  * 2);
    unsigned short* whh1b = (unsigned short*)alloc((size_t)H3 * H  * 2);
    unsigned short* woutb = (unsigned short*)alloc((size_t)V  * H  * 2);
    unsigned short* whidb = (unsigned short*)alloc((size_t)H  * Z  * 2);
    unsigned short* embb  = (unsigned short*)alloc((size_t)V  * E  * 2);
    unsigned short* zbf   = (unsigned short*)alloc((size_t)B  * Z  * 2);
    unsigned short* inp0  = (unsigned short*)alloc((size_t)B  * K0 * 2);
    unsigned short* h0a   = (unsigned short*)alloc((size_t)B  * H  * 2);
    unsigned short* h0b   = (unsigned short*)alloc((size_t)B  * H  * 2);
    unsigned short* h1a   = (unsigned short*)alloc((size_t)B  * H  * 2);
    unsigned short* h1b   = (unsigned short*)alloc((size_t)B  * H  * 2);
    int* tok = (int*)alloc((size_t)B * 4);

    // ---- one-time prep: f32 -> bf16 (weights stay hot in 192MB L2) ----
    f32_to_bf16_kernel<<<1024, 256, 0, stream>>>(W_ih0, wih0b, H3 * K0);
    f32_to_bf16_kernel<<<1024, 256, 0, stream>>>(W_hh0, whh0b, H3 * H);
    f32_to_bf16_kernel<<<1024, 256, 0, stream>>>(W_ih1, wih1b, H3 * H);
    f32_to_bf16_kernel<<<1024, 256, 0, stream>>>(W_hh1, whh1b, H3 * H);
    f32_to_bf16_kernel<<<512,  256, 0, stream>>>(W_out, woutb, V * H);
    f32_to_bf16_kernel<<<512,  256, 0, stream>>>(W_hid, whidb, H * Z);
    f32_to_bf16_kernel<<<512,  256, 0, stream>>>(emb,   embb,  V * E);
    f32_to_bf16_kernel<<<512,  256, 0, stream>>>(z,     zbf,   B * Z);

    // h0 = z @ W_hid^T + b_hid  -> both layer carries start from h0
    dim3 gInit(H / 64, B / 64);
    wmma_gemm_kernel<true><<<gInit, 256, 0, stream>>>(zbf, Z, whidb, Z, b_hid, h0a, H, Z);
    wmma_gemm_kernel<true><<<gInit, 256, 0, stream>>>(zbf, Z, whidb, Z, b_hid, h1a, H, Z);

    zero_f32_kernel<<<512, 256, 0, stream>>>(out, B * V);        // output row 0
    tok_init_kernel<<<2, 256, 0, stream>>>(x, tok);
    concat_z_kernel<<<B, 64, 0, stream>>>(zbf, inp0);            // constant z half
    embed_kernel<<<B, 64, 0, stream>>>(tok, embb, inp0);         // e = emb[tok0]

    unsigned short *h0cur = h0a, *h0nxt = h0b, *h1cur = h1a, *h1nxt = h1b;
    dim3 gGru(H / 64, B / 64);   // 16 x 8
    dim3 gLog(V / 64, B / 64);   //  8 x 8
    for (int s = 1; s < S; ++s) {
        gru_layer_kernel<<<gGru, 256, 0, stream>>>(inp0, h0cur, wih0b, whh0b,
                                                   b_ih0, b_hh0, h0nxt);
        gru_layer_kernel<<<gGru, 256, 0, stream>>>(h0nxt, h1cur, wih1b, whh1b,
                                                   b_ih1, b_hh1, h1nxt);
        wmma_gemm_kernel<false><<<gLog, 256, 0, stream>>>(
            h1nxt, H, woutb, H, b_out, out + (size_t)s * B * V, V, H);
        argmax_kernel<<<B, 256, 0, stream>>>(out + (size_t)s * B * V, tok);
        embed_kernel<<<B, 64, 0, stream>>>(tok, embb, inp0);
        unsigned short* t;
        t = h0cur; h0cur = h0nxt; h0nxt = t;
        t = h1cur; h1cur = h1nxt; h1nxt = t;
    }
}
